// LBANPEncoderLayer_2284922601862
// MI455X (gfx1250) — matrix operand order repaired and checked by reading, not compiled, not run on hardware
//
#include <hip/hip_runtime.h>
#include <hip/hip_bf16.h>
#include <stdint.h>

typedef __attribute__((ext_vector_type(16))) __bf16 v16bf;
typedef __attribute__((ext_vector_type(8)))  __bf16 v8bf;
typedef __attribute__((ext_vector_type(8)))  float  v8f;

// ---------------------------------------------------------------- WMMA core
__device__ __forceinline__ v8f wmma_bf16(v16bf a, v16bf b, v8f c) {
  return __builtin_amdgcn_wmma_f32_16x16x32_bf16(
      false, a, false, b, (short)0, c, false, false);
}

// A-matrix fragment of a 16x32 tile from row-major [rows][ldk] bf16.
// Lane l(0-15): row l, K = {0..7, 16..23}; lane 16+l: row l, K = {8..15, 24..31}.
__device__ __forceinline__ v16bf ld_fragA(const __bf16* p0, int ldk) {
  const int lane = threadIdx.x & 31;
  const int l = lane & 15, hi = lane >> 4;
  const __bf16* p = p0 + (size_t)l * ldk + hi * 8;
  union { v16bf v; v8bf h[2]; } u;
  u.h[0] = *(const v8bf*)(p);
  u.h[1] = *(const v8bf*)(p + 16);
  return u.v;
}

// B-matrix fragment of a 32x16 tile, sourced from B^T stored row-major
// [n][ldk] bf16. Lane l(0-15): col n=l, K = {0..7, 16..23}; lane 16+l: K = {8..15, 24..31}.
__device__ __forceinline__ v16bf ld_fragB(const __bf16* p0, int ldk) {
  const int lane = threadIdx.x & 31;
  const int l = lane & 15, hi = lane >> 4;
  const __bf16* p = p0 + (size_t)l * ldk + hi * 8;
  union { v16bf v; v8bf h[2]; } u;
  u.h[0] = *(const v8bf*)(p);
  u.h[1] = *(const v8bf*)(p + 16);
  return u.v;
}

// ------------------------------------------------- CDNA5 async global->LDS
// GLOBAL_LOAD_ASYNC_TO_LDS_B128 (ASYNCcnt-tracked DMA into LDS, no VGPR data).
__device__ __forceinline__ void async_ld16(__bf16* lds, const __bf16* g) {
  unsigned loff = (unsigned)(uintptr_t)lds;  // LDS aperture: low 32 bits = LDS addr
  asm volatile("global_load_async_to_lds_b128 %0, %1, off"
               :: "v"(loff), "v"((unsigned long long)(uintptr_t)g)
               : "memory");
}
__device__ __forceinline__ void wait_async0() {
  asm volatile("s_wait_asynccnt 0" ::: "memory");
}

// ---------------------------------------------------------------- GEMM
// C[M,N] = alpha*(A[M,K] @ W[K,N]) (+bias[N]) (+res[M,N]); W given TRANSPOSED
// as Wt[N][K] bf16 so both LDS tiles are contiguous 16B chunks (async-copied).
// 256 thr / 8 waves, tile 128x128, k-step 32, double-buffered async pipeline.
template <bool BIAS, bool RES, bool OBF>
__global__ __launch_bounds__(256)
void gemm_bf16_kernel(const __bf16* __restrict__ A, const __bf16* __restrict__ Wt,
                      const float* __restrict__ bias, const float* __restrict__ res,
                      float* __restrict__ outF, __bf16* __restrict__ outB,
                      unsigned M, unsigned N, unsigned K, float alpha) {
  const unsigned tid  = threadIdx.x;
  const unsigned lane = tid & 31;
  const unsigned w    = tid >> 5;
  const unsigned l    = lane & 15, hi = lane >> 4;
  const unsigned waveM = w >> 1, waveN = w & 1;
  const unsigned m0 = blockIdx.y * 128, n0 = blockIdx.x * 128;

  __shared__ __align__(16) __bf16 As[2][128][40];   // [m][k]
  __shared__ __align__(16) __bf16 Bs[2][128][40];   // Wt tile: [n][k]

  // Per-thread staging coords: row = tid>>1, 16 contiguous bf16 at (tid&1)*16.
  const unsigned srow = tid >> 1;
  const unsigned skc  = (tid & 1) * 16;
  const __bf16* gA = A  + (size_t)(m0 + srow) * K + skc;
  const __bf16* gB = Wt + (size_t)(n0 + srow) * K + skc;

  auto issue = [&](unsigned buf, unsigned k0) {
    async_ld16(&As[buf][srow][skc],     gA + k0);
    async_ld16(&As[buf][srow][skc] + 8, gA + k0 + 8);
    async_ld16(&Bs[buf][srow][skc],     gB + k0);
    async_ld16(&Bs[buf][srow][skc] + 8, gB + k0 + 8);
  };

  v8f zero = {};
  v8f acc[2][4];
#pragma unroll
  for (int mt = 0; mt < 2; ++mt)
#pragma unroll
    for (int nt = 0; nt < 4; ++nt) acc[mt][nt] = zero;

  const unsigned nk = K >> 5;
  issue(0, 0);
  for (unsigned ks = 0; ks < nk; ++ks) {
    wait_async0();
    __syncthreads();                       // tile ks resident for all waves
    if (ks + 1 < nk) issue((ks + 1) & 1, (ks + 1) * 32);  // prefetch next

    const unsigned buf = ks & 1;
    v16bf af[2], bfrag[4];
#pragma unroll
    for (int mt = 0; mt < 2; ++mt)
      af[mt] = ld_fragA(&As[buf][waveM * 32 + mt * 16][0], 40);
#pragma unroll
    for (int nt = 0; nt < 4; ++nt)
      bfrag[nt] = ld_fragB(&Bs[buf][waveN * 64 + nt * 16][0], 40);
#pragma unroll
    for (int mt = 0; mt < 2; ++mt)
#pragma unroll
      for (int nt = 0; nt < 4; ++nt)
        acc[mt][nt] = wmma_bf16(af[mt], bfrag[nt], acc[mt][nt]);
    __syncthreads();                       // all waves done reading buf
  }

  // Branch-free epilogue, 32-bit addressing. C: VGPR r -> row hi*8+r, col l.
#pragma unroll
  for (int mt = 0; mt < 2; ++mt)
#pragma unroll
    for (int nt = 0; nt < 4; ++nt)
#pragma unroll
      for (int r = 0; r < 8; ++r) {
        unsigned gr = m0 + waveM * 32 + mt * 16 + hi * 8 + r;
        unsigned gc = n0 + waveN * 64 + nt * 16 + l;
        unsigned idx = gr * N + gc;
        float v = alpha * acc[mt][nt][r];
        if constexpr (BIAS) v += bias[gc];
        if constexpr (RES)  v += res[idx];
        if constexpr (OBF)  outB[idx] = (__bf16)v;
        else                outF[idx] = v;
      }
}

// ---------------------------------------------------------------- Flash attention
// One block per (batch*head, key-split). 8 waves x 16 query rows = 128 queries.
// KV: [B*Nk, 1024] bf16, K at cols [0,512), V at cols [512,1024). Q pre-scaled.
// Writes unnormalized partial O + per-row (m, l) stats; merged by attn_merge.
__global__ __launch_bounds__(256)
void flash_attn_kernel(const __bf16* __restrict__ Q, const __bf16* __restrict__ KV,
                       float* __restrict__ Opart, float* __restrict__ Mst,
                       float* __restrict__ Lst, int Lq, int Nk, int nsplit) {
  const int bh = blockIdx.x;          // 0..63
  const int split = blockIdx.y;
  const int bb = bh >> 3;             // H = 8
  const int hh = bh & 7;
  const int tid  = threadIdx.x;
  const int lane = tid & 31;
  const int w    = tid >> 5;
  const int l    = lane & 15, hi = lane >> 4;

  __shared__ __align__(16) __bf16 Ks[128][72];       // [n][d]
  __shared__ __align__(16) __bf16 Vt[64][136];       // [d][n]
  __shared__ __align__(16) __bf16 Pw[8][16][136];    // per-wave P

  const __bf16* qbp = Q + ((size_t)(bb * Lq + w * 16)) * 512 + hh * 64;
  v16bf qf0 = ld_fragA(qbp, 512);
  v16bf qf1 = ld_fragA(qbp + 32, 512);

  v8f zero = {};
  v8f s[8], o[4];
  float m_run[8], l_run[8];
#pragma unroll
  for (int i = 0; i < 4; ++i) o[i] = zero;
#pragma unroll
  for (int r = 0; r < 8; ++r) { m_run[r] = -3.0e30f; l_run[r] = 0.0f; }

  const int span = Nk / nsplit;
  const int c0 = split * span, c1 = c0 + span;
  const int knr = tid >> 1;                 // staging row 0..127
  const int kc0 = (tid & 1) * 32;           // 32 contiguous bf16

  for (int n0 = c0; n0 < c1; n0 += 128) {
    // K chunk 128x64: async DMA straight into LDS (4 x b128 per thread).
    {
      const __bf16* g = KV + ((size_t)(bb * Nk) + n0 + knr) * 1024 + hh * 64 + kc0;
      __bf16* ld = &Ks[knr][kc0];
#pragma unroll
      for (int j = 0; j < 4; ++j) async_ld16(ld + 8 * j, g + 8 * j);
    }
    // V chunk, transposed into Vt[d][n] (scatter via LDS stores).
    {
      const __bf16* g = KV + ((size_t)(bb * Nk) + n0 + knr) * 1024 + 512 + hh * 64 + kc0;
#pragma unroll
      for (int j = 0; j < 4; ++j) {
        v8bf vv = *(const v8bf*)(g + 8 * j);
#pragma unroll
        for (int e = 0; e < 8; ++e) Vt[kc0 + 8 * j + e][knr] = vv[e];
      }
    }
    wait_async0();
    __syncthreads();

    // S = Q K^T : 8 column tiles of 16, contraction dh=64 in 2 steps.
#pragma unroll
    for (int nt = 0; nt < 8; ++nt) {
      v8f a = zero;
      a = wmma_bf16(qf0, ld_fragB(&Ks[nt * 16][0], 72), a);
      a = wmma_bf16(qf1, ld_fragB(&Ks[nt * 16][32], 72), a);
      s[nt] = a;
    }

    // Streaming softmax per row (row = hi*8 + r shared by a 16-lane half).
    float mnew[8], corr[8];
#pragma unroll
    for (int r = 0; r < 8; ++r) {
      float mv = s[0][r];
#pragma unroll
      for (int nt = 1; nt < 8; ++nt) mv = fmaxf(mv, s[nt][r]);
      for (int msk = 8; msk >= 1; msk >>= 1)
        mv = fmaxf(mv, __shfl_xor(mv, msk, 32));
      mnew[r] = fmaxf(m_run[r], mv);
      corr[r] = __expf(m_run[r] - mnew[r]);
      float rs = 0.0f;
#pragma unroll
      for (int nt = 0; nt < 8; ++nt) {
        float p = __expf(s[nt][r] - mnew[r]);
        s[nt][r] = p;
        rs += p;
      }
      for (int msk = 8; msk >= 1; msk >>= 1) rs += __shfl_xor(rs, msk, 32);
      l_run[r] = l_run[r] * corr[r] + rs;
      m_run[r] = mnew[r];
    }
#pragma unroll
    for (int dt = 0; dt < 4; ++dt)
#pragma unroll
      for (int r = 0; r < 8; ++r) o[dt][r] *= corr[r];

    // P -> LDS (wave-private), bf16.
#pragma unroll
    for (int nt = 0; nt < 8; ++nt)
#pragma unroll
      for (int r = 0; r < 8; ++r)
        Pw[w][hi * 8 + r][nt * 16 + l] = (__bf16)s[nt][r];
    __syncthreads();

    // O += P V : contraction over 128 keys in 4 steps of 32.
#pragma unroll
    for (int kn = 0; kn < 128; kn += 32) {
      v16bf pa = ld_fragA(&Pw[w][0][kn], 136);
#pragma unroll
      for (int dt = 0; dt < 4; ++dt)
        o[dt] = wmma_bf16(pa, ld_fragB(&Vt[dt * 16][kn], 136), o[dt]);
    }
    __syncthreads();
  }

  // Store unnormalized partial O and per-row stats for the merge pass.
  const unsigned pbase = ((unsigned)(split * 64 + bh) * 128 + w * 16 + hi * 8) * 64;
#pragma unroll
  for (int dt = 0; dt < 4; ++dt)
#pragma unroll
    for (int r = 0; r < 8; ++r)
      Opart[pbase + (unsigned)r * 64 + dt * 16 + l] = o[dt][r];
  if (l == 0) {
#pragma unroll
    for (int r = 0; r < 8; ++r) {
      unsigned sidx = (unsigned)(split * 64 + bh) * 128 + w * 16 + hi * 8 + r;
      Mst[sidx] = m_run[r];
      Lst[sidx] = l_run[r];
    }
  }
}

// Merge key-splits: softmax-weighted recombination -> bf16 [B*L, 512].
__global__ __launch_bounds__(256)
void attn_merge_kernel(const float* __restrict__ Opart, const float* __restrict__ Mst,
                       const float* __restrict__ Lst, __bf16* __restrict__ O,
                       int nsplit) {
  unsigned i = blockIdx.x * 256 + threadIdx.x;     // over 64*128*64
  unsigned d   = i & 63;
  unsigned row = (i >> 6) & 127;
  unsigned bh  = i >> 13;
  unsigned bb = bh >> 3, hh = bh & 7;
  float mmax = -3.0e30f;
  for (int sp = 0; sp < nsplit; ++sp)
    mmax = fmaxf(mmax, Mst[(unsigned)(sp * 64 + bh) * 128 + row]);
  float num = 0.0f, den = 0.0f;
  for (int sp = 0; sp < nsplit; ++sp) {
    unsigned sidx = (unsigned)(sp * 64 + bh) * 128 + row;
    float sc = __expf(Mst[sidx] - mmax);
    den += sc * Lst[sidx];
    num += sc * Opart[sidx * 64 + d];
  }
  O[((bb * 128 + row) * 512u) + hh * 64 + d] = (__bf16)(num / den);
}

// ------------------------------------------- f32 [K][N] -> bf16 transpose [N][K]
__global__ __launch_bounds__(256)
void transpose_bf16_kernel(const float* __restrict__ in, __bf16* __restrict__ out,
                           int K, int N) {
  __shared__ float t[32][33];
  const int kb = blockIdx.y * 32, nb = blockIdx.x * 32;
  const int tx = threadIdx.x & 31, ty = threadIdx.x >> 5;   // 32 x 8
#pragma unroll
  for (int i = 0; i < 32; i += 8)
    t[ty + i][tx] = in[(size_t)(kb + ty + i) * N + nb + tx];
  __syncthreads();
#pragma unroll
  for (int i = 0; i < 32; i += 8)
    out[(size_t)(nb + ty + i) * K + kb + tx] = (__bf16)t[tx][ty + i];
}

// ---------------------------------------------------------------- LayerNorm -> bf16
__global__ __launch_bounds__(256)
void layernorm_bf16_kernel(const float* __restrict__ x, const float* __restrict__ g,
                           const float* __restrict__ b, __bf16* __restrict__ out) {
  const int row = blockIdx.x;
  const int tid = threadIdx.x;
  const float* xr = x + (size_t)row * 512;
  float v0 = xr[tid], v1 = xr[tid + 256];
  float s = v0 + v1, ss = v0 * v0 + v1 * v1;
  for (int m = 16; m >= 1; m >>= 1) {
    s  += __shfl_xor(s, m, 32);
    ss += __shfl_xor(ss, m, 32);
  }
  __shared__ float r0[8], r1[8];
  if ((tid & 31) == 0) { r0[tid >> 5] = s; r1[tid >> 5] = ss; }
  __syncthreads();
  float S = 0.0f, SS = 0.0f;
#pragma unroll
  for (int i = 0; i < 8; ++i) { S += r0[i]; SS += r1[i]; }
  float mu  = S * (1.0f / 512.0f);
  float var = SS * (1.0f / 512.0f) - mu * mu;
  float inv = rsqrtf(var + 1e-5f);
  out[(size_t)row * 512 + tid]       = (__bf16)((v0 - mu) * inv * g[tid] + b[tid]);
  out[(size_t)row * 512 + tid + 256] = (__bf16)((v1 - mu) * inv * g[tid + 256] + b[tid + 256]);
}

// ---------------------------------------------------------------- GEGLU
__global__ __launch_bounds__(256)
void geglu_kernel(const float* __restrict__ h, __bf16* __restrict__ out) {
  unsigned i   = blockIdx.x * 256 + threadIdx.x;
  unsigned row = i >> 11;
  unsigned col = i & 2047;
  float a = h[row * 4096u + col];
  float g = h[row * 4096u + 2048u + col];
  float gelu = 0.5f * g * (1.0f + erff(g * 0.70710678118f));
  out[i] = (__bf16)(a * gelu);
}

// ---------------------------------------------------------------- host orchestration
extern "C" void kernel_launch(void* const* d_in, const int* in_sizes, int n_in,
                              void* d_out, int out_size, void* d_ws, size_t ws_size,
                              hipStream_t stream) {
  (void)in_sizes; (void)n_in; (void)out_size; (void)ws_size;
  constexpr int Bz = 8, L = 128, N = 4096, D = 512, FF = 2048;
  constexpr int ML = Bz * L;        // 1024
  constexpr int MN = Bz * N;        // 32768
  constexpr float SCALE = 0.125f;   // (512/8)^-0.5

  const float* context = (const float*)d_in[0];
  const float* latents = (const float*)d_in[1];
  const float* ca_ln_w = (const float*)d_in[2];
  const float* ca_ln_b = (const float*)d_in[3];
  const float* ca_lnc_w = (const float*)d_in[4];
  const float* ca_lnc_b = (const float*)d_in[5];
  const float* ca_wq = (const float*)d_in[6];
  const float* ca_wkv = (const float*)d_in[7];
  const float* ca_wo = (const float*)d_in[8];
  const float* ca_bo = (const float*)d_in[9];
  const float* cf_ln_w = (const float*)d_in[10];
  const float* cf_ln_b = (const float*)d_in[11];
  const float* cf_w1 = (const float*)d_in[12];
  const float* cf_b1 = (const float*)d_in[13];
  const float* cf_w2 = (const float*)d_in[14];
  const float* cf_b2 = (const float*)d_in[15];
  const float* sa_ln_w = (const float*)d_in[16];
  const float* sa_ln_b = (const float*)d_in[17];
  const float* sa_wq = (const float*)d_in[18];
  const float* sa_wkv = (const float*)d_in[19];
  const float* sa_wo = (const float*)d_in[20];
  const float* sa_bo = (const float*)d_in[21];
  const float* lf_ln_w = (const float*)d_in[22];
  const float* lf_ln_b = (const float*)d_in[23];
  const float* lf_w1 = (const float*)d_in[24];
  const float* lf_b1 = (const float*)d_in[25];
  const float* lf_w2 = (const float*)d_in[26];
  const float* lf_b2 = (const float*)d_in[27];

  char* ws = (char*)d_ws;
  size_t off = 0;
  auto take = [&](size_t bytes) -> char* {
    char* p = ws + off;
    off = (off + bytes + 255) & ~(size_t)255;
    return p;
  };

  // Transposed bf16 weights Wt[N][K].
  __bf16* wq1  = (__bf16*)take((size_t)D * D * 2);
  __bf16* wkv1 = (__bf16*)take((size_t)D * 2 * D * 2);
  __bf16* wo1  = (__bf16*)take((size_t)D * D * 2);
  __bf16* w1c  = (__bf16*)take((size_t)D * 2 * FF * 2);
  __bf16* w2c  = (__bf16*)take((size_t)FF * D * 2);
  __bf16* wq2  = (__bf16*)take((size_t)D * D * 2);
  __bf16* wkv2 = (__bf16*)take((size_t)D * 2 * D * 2);
  __bf16* wo2  = (__bf16*)take((size_t)D * D * 2);
  __bf16* w1l  = (__bf16*)take((size_t)D * 2 * FF * 2);
  __bf16* w2l  = (__bf16*)take((size_t)FF * D * 2);
  __bf16* xn   = (__bf16*)take((size_t)ML * D * 2);
  __bf16* cn   = (__bf16*)take((size_t)MN * D * 2);
  __bf16* qb   = (__bf16*)take((size_t)ML * D * 2);
  __bf16* kvb  = (__bf16*)take((size_t)MN * 2 * D * 2);
  __bf16* aob  = (__bf16*)take((size_t)ML * D * 2);
  float*  hbuf = (float*)take((size_t)ML * 2 * FF * 4);
  __bf16* ggb  = (__bf16*)take((size_t)ML * FF * 2);
  float*  x1   = (float*)take((size_t)ML * D * 4);
  float*  x2   = (float*)take((size_t)ML * D * 4);
  float*  x3   = (float*)take((size_t)ML * D * 4);
  float*  opart = (float*)take((size_t)4 * 64 * 128 * 64 * 4);
  float*  mst   = (float*)take((size_t)4 * 64 * 128 * 4);
  float*  lst   = (float*)take((size_t)4 * 64 * 128 * 4);
  float*  xout = (float*)d_out;

  auto tr = [&](const float* wsrc, __bf16* wt, int Kd, int Nd) {
    transpose_bf16_kernel<<<dim3(Nd / 32, Kd / 32), 256, 0, stream>>>(wsrc, wt, Kd, Nd);
  };
  auto gemmB = [&](const __bf16* A, const __bf16* Wt, __bf16* oB,
                   unsigned M, unsigned Nn, unsigned K, float alpha) {
    gemm_bf16_kernel<false, false, true><<<dim3(Nn / 128, M / 128), 256, 0, stream>>>(
        A, Wt, nullptr, nullptr, nullptr, oB, M, Nn, K, alpha);
  };
  auto gemmBiasF = [&](const __bf16* A, const __bf16* Wt, const float* bias, float* oF,
                       unsigned M, unsigned Nn, unsigned K) {
    gemm_bf16_kernel<true, false, false><<<dim3(Nn / 128, M / 128), 256, 0, stream>>>(
        A, Wt, bias, nullptr, oF, nullptr, M, Nn, K, 1.0f);
  };
  auto gemmBiasResF = [&](const __bf16* A, const __bf16* Wt, const float* bias,
                          const float* res, float* oF,
                          unsigned M, unsigned Nn, unsigned K) {
    gemm_bf16_kernel<true, true, false><<<dim3(Nn / 128, M / 128), 256, 0, stream>>>(
        A, Wt, bias, res, oF, nullptr, M, Nn, K, 1.0f);
  };

  // Weights -> transposed bf16 (one-time per call; feeds async tile copies).
  tr(ca_wq, wq1, D, D);
  tr(ca_wkv, wkv1, D, 2 * D);
  tr(ca_wo, wo1, D, D);
  tr(cf_w1, w1c, D, 2 * FF);
  tr(cf_w2, w2c, FF, D);
  tr(sa_wq, wq2, D, D);
  tr(sa_wkv, wkv2, D, 2 * D);
  tr(sa_wo, wo2, D, D);
  tr(lf_w1, w1l, D, 2 * FF);
  tr(lf_w2, w2l, FF, D);

  // ---- cross attention ----
  layernorm_bf16_kernel<<<ML, 256, 0, stream>>>(latents, ca_ln_w, ca_ln_b, xn);
  layernorm_bf16_kernel<<<MN, 256, 0, stream>>>(context, ca_lnc_w, ca_lnc_b, cn);
  gemmB(xn, wq1, qb, ML, D, D, SCALE);            // Q (pre-scaled)
  gemmB(cn, wkv1, kvb, MN, 2 * D, D, 1.0f);       // KV
  flash_attn_kernel<<<dim3(64, 4), 256, 0, stream>>>(qb, kvb, opart, mst, lst, L, N, 4);
  attn_merge_kernel<<<2048, 256, 0, stream>>>(opart, mst, lst, aob, 4);
  gemmBiasResF(aob, wo1, ca_bo, latents, x1, ML, D, D);

  // ---- cross FFN (GEGLU) ----
  layernorm_bf16_kernel<<<ML, 256, 0, stream>>>(x1, cf_ln_w, cf_ln_b, xn);
  gemmBiasF(xn, w1c, cf_b1, hbuf, ML, 2 * FF, D);
  geglu_kernel<<<(ML * FF) / 256, 256, 0, stream>>>(hbuf, ggb);
  gemmBiasResF(ggb, w2c, cf_b2, x1, x2, ML, D, FF);

  // ---- latent self attention ----
  layernorm_bf16_kernel<<<ML, 256, 0, stream>>>(x2, sa_ln_w, sa_ln_b, xn);
  gemmB(xn, wq2, qb, ML, D, D, SCALE);
  gemmB(xn, wkv2, kvb, ML, 2 * D, D, 1.0f);
  flash_attn_kernel<<<dim3(64, 1), 256, 0, stream>>>(qb, kvb, opart, mst, lst, L, L, 1);
  attn_merge_kernel<<<2048, 256, 0, stream>>>(opart, mst, lst, aob, 1);
  gemmBiasResF(aob, wo2, sa_bo, x2, x3, ML, D, D);

  // ---- latent FFN (GEGLU) ----
  layernorm_bf16_kernel<<<ML, 256, 0, stream>>>(x3, lf_ln_w, lf_ln_b, xn);
  gemmBiasF(xn, w1l, lf_b1, hbuf, ML, 2 * FF, D);
  geglu_kernel<<<(ML * FF) / 256, 256, 0, stream>>>(hbuf, ggb);
  gemmBiasResF(ggb, w2l, lf_b2, x3, xout, ML, D, FF);
}
